// ChannelAttention_38491496907349
// MI455X (gfx1250) — matrix operand order repaired and checked by reading, not compiled, not run on hardware
//
#include <hip/hip_runtime.h>

// ---------- types ----------
typedef __attribute__((ext_vector_type(16))) _Float16 v16h;
typedef __attribute__((ext_vector_type(8)))  _Float16 v8h;
typedef __attribute__((ext_vector_type(8)))  float    v8f;
typedef __attribute__((ext_vector_type(4)))  float    f4;
typedef __attribute__((ext_vector_type(4)))  int      i4;

__device__ __forceinline__ v16h ld_frag2(const _Float16* p0, const _Float16* p1) {
  union { v8h h[2]; v16h v; } u;
  u.h[0] = *(const v8h*)p0;
  u.h[1] = *(const v8h*)p1;
  return u.v;
}
__device__ __forceinline__ v16h ld_frag32(const _Float16* p) { return *(const v16h*)p; }

__device__ __forceinline__ v8f wmma32(v16h a, v16h b, v8f c) {
  // D = A(16x32 f16) * B(32x16 f16) + C(16x16 f32)
  return __builtin_amdgcn_wmma_f32_16x16x32_f16(false, a, false, b, (short)0, c, false, false);
}

// ---------- async global->LDS copy (CDNA5 ASYNCcnt path), compile-safe fallback ----------
#if defined(__gfx1250__) && __has_builtin(__builtin_amdgcn_global_load_async_to_lds_b128)
#define ATHENA_ASYNC_LDS 1
typedef __attribute__((address_space(1))) i4 gi4_t;
typedef __attribute__((address_space(3))) i4 li4_t;
#endif

__device__ __forceinline__ void async_copy16(const void* g, void* l) {
#ifdef ATHENA_ASYNC_LDS
  __builtin_amdgcn_global_load_async_to_lds_b128(
      (gi4_t*)(unsigned long long)(uintptr_t)g,
      (li4_t*)(unsigned)(uintptr_t)l, 0, 0);
#else
  *(v8h*)l = *(const v8h*)g;
#endif
}
__device__ __forceinline__ void async_wait0() {
#ifdef ATHENA_ASYNC_LDS
#if __has_builtin(__builtin_amdgcn_s_wait_asynccnt)
  __builtin_amdgcn_s_wait_asynccnt(0);
#else
  asm volatile("s_wait_asynccnt 0" ::: "memory");
#endif
#endif
}

// ---------- tiny utility kernels ----------
__global__ void k_zero(float* p, int n) {
  int i = blockIdx.x * blockDim.x + threadIdx.x;
  if (i < n) p[i] = 0.f;
}

// pack w1 [64oc][256ic] -> [s8][t4][lane32][16]  (B-frag order)
__global__ void k_pack_1x1(const float* __restrict__ w, _Float16* __restrict__ dst) {
  const int total = 8 * 4 * 32 * 16;
  for (int i = blockIdx.x * blockDim.x + threadIdx.x; i < total; i += gridDim.x * blockDim.x) {
    int j = i & 15, lane = (i >> 4) & 31, t = (i >> 9) & 3, s = i >> 11;
    int n = t * 16 + (lane & 15);             // oc
    int k = s * 32 + ((lane >> 4) * 8) + (j & 7) + ((j >> 3) * 16);  // ic
    dst[i] = (_Float16)w[n * 256 + k];
  }
}

// pack wK [64oc][64ic][K][K] -> [pos(K*K)][s2][t4][lane32][16]
__global__ void k_pack_conv(const float* __restrict__ w, _Float16* __restrict__ dst, int K) {
  const int total = K * K * 4096;
  for (int i = blockIdx.x * blockDim.x + threadIdx.x; i < total; i += gridDim.x * blockDim.x) {
    int j = i & 15, lane = (i >> 4) & 31, t = (i >> 9) & 3, s = (i >> 11) & 1, pos = i >> 12;
    int dy = pos / K, dx = pos % K;
    int n = t * 16 + (lane & 15);             // oc
    int k = s * 32 + ((lane >> 4) * 8) + (j & 7) + ((j >> 3) * 16);  // ic
    dst[i] = (_Float16)w[((n * 64 + k) * K + dy) * K + dx];
  }
}

// pack fw1 [16cr][256c] -> [s8][lane32][16]
__global__ void k_pack_fw1(const float* __restrict__ w, _Float16* __restrict__ dst) {
  const int total = 8 * 32 * 16;
  for (int i = blockIdx.x * blockDim.x + threadIdx.x; i < total; i += gridDim.x * blockDim.x) {
    int j = i & 15, lane = (i >> 4) & 31, s = i >> 9;
    int n = lane & 15;
    int k = s * 32 + ((lane >> 4) * 8) + (j & 7) + ((j >> 3) * 16);
    dst[i] = (_Float16)w[n * 256 + k];
  }
}

// pack fw2 [256oc][16cr] -> [t16][lane32][16] with K padded 16->32 (zeros)
__global__ void k_pack_fw2(const float* __restrict__ w, _Float16* __restrict__ dst) {
  const int total = 16 * 32 * 16;
  for (int i = blockIdx.x * blockDim.x + threadIdx.x; i < total; i += gridDim.x * blockDim.x) {
    int j = i & 15, lane = (i >> 4) & 31, t = i >> 9;
    int n = t * 16 + (lane & 15);
    int kk = ((lane >> 4) * 8) + (j & 7) + ((j >> 3) * 16);
    dst[i] = (kk < 16) ? (_Float16)w[n * 16 + kk] : (_Float16)0.f;
  }
}

// per-(b,c) max & mean of x over HW
__global__ __launch_bounds__(256) void k_reduce(const float* __restrict__ x,
                                                float* __restrict__ maxr, float* __restrict__ avgr) {
  __shared__ float smax[256], ssum[256];
  int tid = threadIdx.x, bc = blockIdx.x;
  const float* p = x + (size_t)bc * 4096;
  float mx = -3.4e38f, sm = 0.f;
  for (int i = tid; i < 4096; i += 256) { float v = p[i]; mx = fmaxf(mx, v); sm += v; }
  smax[tid] = mx; ssum[tid] = sm;
  __syncthreads();
  for (int o = 128; o > 0; o >>= 1) {
    if (tid < o) { smax[tid] = fmaxf(smax[tid], smax[tid + o]); ssum[tid] += ssum[tid + o]; }
    __syncthreads();
  }
  if (tid == 0) { maxr[bc] = smax[0]; avgr[bc] = ssum[0] * (1.0f / 4096.0f); }
}

// ---------- conv1x1 via WMMA, 2 oc-tiles/wave: y16[b][pix][oc] (NHWC f16) ----------
__global__ __launch_bounds__(256) void k_conv1x1(const float* __restrict__ x,
                                                 const _Float16* __restrict__ wp,
                                                 const float* __restrict__ b1,
                                                 _Float16* __restrict__ y16) {
  __shared__ __attribute__((aligned(16))) _Float16 ax[64 * 256];  // 64 pixels x 256 ch (32KB)
  int tid = threadIdx.x;
  int b = blockIdx.x >> 6;
  int pix0 = (blockIdx.x & 63) * 64;
  const float* xb = x + (size_t)b * 256 * 4096;
  for (int idx = tid; idx < 256 * 16; idx += 256) {
    int c = idx >> 4, pg = idx & 15;
    f4 v = *(const f4*)(xb + (size_t)c * 4096 + pix0 + pg * 4);
    int pxl = pg * 4;
    ax[(pxl + 0) * 256 + c] = (_Float16)v.x;
    ax[(pxl + 1) * 256 + c] = (_Float16)v.y;
    ax[(pxl + 2) * 256 + c] = (_Float16)v.z;
    ax[(pxl + 3) * 256 + c] = (_Float16)v.w;
  }
  __syncthreads();
  int wave = tid >> 5, lane = tid & 31;
  int strip = wave >> 1, tp = wave & 1;
  int m = lane & 15, lh = lane >> 4, k0 = lh * 8;
  v8f acc0 = {}, acc1 = {};
  const _Float16* arow = &ax[(strip * 16 + m) * 256];
  const _Float16* wbase = wp + (size_t)tp * 1024 + lane * 16;
  for (int s = 0; s < 8; ++s) {
    v16h a = ld_frag2(arow + s * 32 + k0, arow + s * 32 + k0 + 16);
    v16h bm0 = ld_frag32(wbase + (size_t)s * 2048);
    v16h bm1 = ld_frag32(wbase + (size_t)s * 2048 + 512);
    acc0 = wmma32(a, bm0, acc0);
    acc1 = wmma32(a, bm1, acc1);
  }
  int oc0 = tp * 32 + m, oc1 = oc0 + 16;
  float bias0 = b1[oc0], bias1 = b1[oc1];
  _Float16* yb = y16 + ((size_t)b * 4096 + pix0 + strip * 16 + lh * 8) * 64;
#pragma unroll
  for (int v = 0; v < 8; ++v) {
    yb[(size_t)v * 64 + oc0] = (_Float16)(acc0[v] + bias0);
    yb[(size_t)v * 64 + oc1] = (_Float16)(acc1[v] + bias1);
  }
}

// ---------- conv KxK implicit-GEMM WMMA, async LDS staging, 2 oc-tiles/wave ----------
template <int K>
__global__ __launch_bounds__(256) void k_conv(const _Float16* __restrict__ y16,
                                              const _Float16* __restrict__ wp,
                                              const float* __restrict__ bias,
                                              float* __restrict__ cat, int chbase,
                                              float* __restrict__ stat) {
  constexpr int R = K / 2;
  constexpr int ROWS = 4 + 2 * R;
  constexpr int COLS = 16 + 2 * R;
  __shared__ __attribute__((aligned(16))) _Float16 tile[ROWS * COLS * 64];
  __shared__ float lsum[64], lsum2[64];
  int tid = threadIdx.x;
  if (tid < 64) { lsum[tid] = 0.f; lsum2[tid] = 0.f; }
  int b = blockIdx.x >> 6, rem = blockIdx.x & 63;
  int h0 = (rem >> 2) * 4, w0 = (rem & 3) * 16;
  // stage halo tile: async DMA for in-bounds, LDS zero-stores for the padding
  for (int idx = tid; idx < ROWS * COLS * 8; idx += 256) {
    int cc = idx & 7, col = (idx >> 3) % COLS, row = (idx >> 3) / COLS;
    int gh = h0 - R + row, gw = w0 - R + col;
    _Float16* ldst = tile + (size_t)idx * 8;
    if ((unsigned)gh < 64u && (unsigned)gw < 64u) {
      async_copy16(y16 + (((size_t)b * 4096 + gh * 64 + gw) * 64 + cc * 8), ldst);
    } else {
      v8h z = {};
      *(v8h*)ldst = z;
    }
  }
  async_wait0();
  __syncthreads();
  int wave = tid >> 5, lane = tid & 31;
  int wrow = wave >> 1, tp = wave & 1;
  int m = lane & 15, lh = lane >> 4, k0 = lh * 8;
  int oc0 = tp * 32 + m, oc1 = oc0 + 16;
  float b0 = bias[oc0], b1v = bias[oc1];
  v8f acc0 = {b0, b0, b0, b0, b0, b0, b0, b0};
  v8f acc1 = {b1v, b1v, b1v, b1v, b1v, b1v, b1v, b1v};
  for (int dy = 0; dy < K; ++dy)
    for (int dx = 0; dx < K; ++dx) {
      const _Float16* arow = tile + ((size_t)((wrow + dy) * COLS + (m + dx))) * 64;
      const _Float16* wb = wp + (size_t)(dy * K + dx) * 4096 + (size_t)tp * 1024 + lane * 16;
#pragma unroll
      for (int s = 0; s < 2; ++s) {
        v16h a = ld_frag2(arow + s * 32 + k0, arow + s * 32 + k0 + 16);
        v16h bm0 = ld_frag32(wb + s * 2048);
        v16h bm1 = ld_frag32(wb + s * 2048 + 512);
        acc0 = wmma32(a, bm0, acc0);
        acc1 = wmma32(a, bm1, acc1);
      }
    }
  int h = h0 + wrow, wc0 = w0 + lh * 8;
  size_t base = (((size_t)b * 256 + chbase) * 64 + h) * 64 + wc0;
  float* dst0 = cat + base + (size_t)oc0 * 4096;
  float* dst1 = cat + base + (size_t)oc1 * 4096;
  float s1a = 0.f, s2a = 0.f, s1b = 0.f, s2b = 0.f;
#pragma unroll
  for (int v = 0; v < 8; ++v) {
    float z0 = acc0[v]; s1a += z0; s2a += z0 * z0;
    float z1 = acc1[v]; s1b += z1; s2b += z1 * z1;
  }
  f4 a00 = {acc0[0], acc0[1], acc0[2], acc0[3]};
  f4 a01 = {acc0[4], acc0[5], acc0[6], acc0[7]};
  f4 a10 = {acc1[0], acc1[1], acc1[2], acc1[3]};
  f4 a11 = {acc1[4], acc1[5], acc1[6], acc1[7]};
  *(f4*)dst0 = a00; *(f4*)(dst0 + 4) = a01;
  *(f4*)dst1 = a10; *(f4*)(dst1 + 4) = a11;
  atomicAdd(&lsum[oc0], s1a);  atomicAdd(&lsum2[oc0], s2a);   // LDS atomics
  atomicAdd(&lsum[oc1], s1b);  atomicAdd(&lsum2[oc1], s2b);
  __syncthreads();
  if (tid < 64) {
    atomicAdd(&stat[chbase + tid], lsum[tid]);
    atomicAdd(&stat[192 + chbase + tid], lsum2[tid]);
  }
}

// ---------- maxpool 2x2 (y16 NHWC) -> pool[b][32][32][64] f16 ----------
__global__ __launch_bounds__(256) void k_pool(const _Float16* __restrict__ y16, _Float16* __restrict__ pool) {
  int idx = blockIdx.x * blockDim.x + threadIdx.x;  // 65536
  int c8 = idx & 7, pw = (idx >> 3) & 31, ph = (idx >> 8) & 31, b = idx >> 13;
  const _Float16* p00 = y16 + (((size_t)b * 4096 + (2 * ph) * 64 + 2 * pw) * 64 + c8 * 8);
  const _Float16* p01 = p00 + 64;
  const _Float16* p10 = p00 + 64 * 64;
  const _Float16* p11 = p10 + 64;
  _Float16* d = pool + (((size_t)b * 1024 + ph * 32 + pw) * 64 + c8 * 8);
#pragma unroll
  for (int j = 0; j < 8; ++j) {
    float mx = fmaxf(fmaxf((float)p00[j], (float)p01[j]), fmaxf((float)p10[j], (float)p11[j]));
    d[j] = (_Float16)mx;
  }
}

// ---------- bilinear x2 upsample (align_corners=False) + relu -> cat ch 192..255 ----------
__global__ __launch_bounds__(256) void k_bilinear(const _Float16* __restrict__ pool, float* __restrict__ cat) {
  int idx = blockIdx.x * blockDim.x + threadIdx.x;  // 262144
  int c8 = idx & 7, w = (idx >> 3) & 63, h = (idx >> 9) & 63, b = idx >> 15;
  float sh = h * 0.5f - 0.25f, sw = w * 0.5f - 0.25f;
  int ih0 = (int)floorf(sh), iw0 = (int)floorf(sw);
  float fh = sh - ih0, fw = sw - iw0;
  int ih1 = min(max(ih0 + 1, 0), 31), iw1 = min(max(iw0 + 1, 0), 31);
  ih0 = min(max(ih0, 0), 31); iw0 = min(max(iw0, 0), 31);
  const _Float16* base = pool + (size_t)b * 1024 * 64;
  const _Float16* p00 = base + ((ih0 * 32 + iw0) * 64 + c8 * 8);
  const _Float16* p01 = base + ((ih0 * 32 + iw1) * 64 + c8 * 8);
  const _Float16* p10 = base + ((ih1 * 32 + iw0) * 64 + c8 * 8);
  const _Float16* p11 = base + ((ih1 * 32 + iw1) * 64 + c8 * 8);
#pragma unroll
  for (int j = 0; j < 8; ++j) {
    float v0 = (1.f - fw) * (float)p00[j] + fw * (float)p01[j];
    float v1 = (1.f - fw) * (float)p10[j] + fw * (float)p11[j];
    float v = (1.f - fh) * v0 + fh * v1;
    v = fmaxf(v, 0.f);
    int c = c8 * 8 + j;
    cat[(((size_t)b * 256 + 192 + c) * 4096) + h * 64 + w] = v;
  }
}

// ---------- BN (training, biased var over N*H*W=32768) + relu, in-place on cat ch 0..191 ----------
__global__ __launch_bounds__(256) void k_bnrelu(float* __restrict__ cat, const float* __restrict__ stat,
                                                const float* g2, const float* bt2,
                                                const float* g3, const float* bt3,
                                                const float* g4, const float* bt4) {
  int idx = blockIdx.x * blockDim.x + threadIdx.x;  // 6291456
  int pix = idx & 4095;
  int bc = idx >> 12;              // 0..1535
  int ch = bc % 192, b = bc / 192;
  const float invN = 1.0f / 32768.0f;
  float mean = stat[ch] * invN;
  float var = stat[192 + ch] * invN - mean * mean;
  float inv = rsqrtf(var + 1e-5f);
  int br = ch >> 6, cl = ch & 63;
  float g, bt;
  if (br == 0)      { g = g2[cl]; bt = bt2[cl]; }
  else if (br == 1) { g = g3[cl]; bt = bt3[cl]; }
  else              { g = g4[cl]; bt = bt4[cl]; }
  size_t off = (((size_t)b * 256 + ch) * 4096) + pix;
  float v = (cat[off] - mean) * inv * g + bt;
  cat[off] = fmaxf(v, 0.f);
}

// ---------- 3x3 median (reflect pad) on cat -> med16[b][pix][c] NHWC f16 ----------
__device__ __forceinline__ void mm_sort2(float& a, float& b) {
  float t = fminf(a, b); b = fmaxf(a, b); a = t;
}
__global__ __launch_bounds__(256) void k_median(const float* __restrict__ cat, _Float16* __restrict__ med16) {
  int idx = blockIdx.x * blockDim.x + threadIdx.x;  // 8388608
  int w = idx & 63, h = (idx >> 6) & 63, c = (idx >> 12) & 255, b = idx >> 20;
  const float* p = cat + ((size_t)b * 256 + c) * 4096;
  int hm = (h == 0) ? 1 : h - 1, hp = (h == 63) ? 62 : h + 1;
  int wm = (w == 0) ? 1 : w - 1, wp = (w == 63) ? 62 : w + 1;
  float p0 = p[hm * 64 + wm], p1 = p[hm * 64 + w], p2 = p[hm * 64 + wp];
  float p3 = p[h  * 64 + wm], p4 = p[h  * 64 + w], p5 = p[h  * 64 + wp];
  float p6 = p[hp * 64 + wm], p7 = p[hp * 64 + w], p8 = p[hp * 64 + wp];
  mm_sort2(p1, p2); mm_sort2(p4, p5); mm_sort2(p7, p8);
  mm_sort2(p0, p1); mm_sort2(p3, p4); mm_sort2(p6, p7);
  mm_sort2(p1, p2); mm_sort2(p4, p5); mm_sort2(p7, p8);
  mm_sort2(p0, p3); mm_sort2(p5, p8); mm_sort2(p4, p7);
  mm_sort2(p3, p6); mm_sort2(p1, p4); mm_sort2(p2, p5);
  mm_sort2(p4, p7); mm_sort2(p4, p2); mm_sort2(p6, p4);
  mm_sort2(p4, p2);
  med16[((size_t)b * 4096 + h * 64 + w) * 256 + c] = (_Float16)p4;
}

// ---------- fc(max_r)+fc(avg_r) -> rvec[b][256] ----------
__global__ __launch_bounds__(256) void k_rvec(const float* __restrict__ maxr, const float* __restrict__ avgr,
                                              const float* __restrict__ fw1, const float* __restrict__ fb1,
                                              const float* __restrict__ fw2, const float* __restrict__ fb2,
                                              float* __restrict__ rvec) {
  __shared__ float h1m[16], h1a[16];
  int b = blockIdx.x, tid = threadIdx.x;
  if (tid < 16) {
    float dm = 0.f, da = 0.f;
    for (int c = 0; c < 256; ++c) {
      float wv = fw1[tid * 256 + c];
      dm += wv * maxr[b * 256 + c];
      da += wv * avgr[b * 256 + c];
    }
    h1m[tid] = fmaxf(dm + fb1[tid], 0.f);
    h1a[tid] = fmaxf(da + fb1[tid], 0.f);
  }
  __syncthreads();
  float acc = 2.f * fb2[tid];
#pragma unroll
  for (int j = 0; j < 16; ++j) acc += fw2[tid * 16 + j] * (h1m[j] + h1a[j]);
  rvec[b * 256 + tid] = acc;
}

// ---------- fc block on med via 2 WMMA GEMMs + sigmoid -> out (NCHW f32) ----------
__global__ __launch_bounds__(256) void k_fc(const _Float16* __restrict__ med16,
                                            const _Float16* __restrict__ wpf1,
                                            const _Float16* __restrict__ wpf2,
                                            const float* __restrict__ fb1,
                                            const float* __restrict__ fb2,
                                            const float* __restrict__ rv,
                                            float* __restrict__ out) {
  __shared__ __attribute__((aligned(16))) _Float16 hbuf[8][16][32];
  int tid = threadIdx.x, wave = tid >> 5, lane = tid & 31;
  int m = lane & 15, lh = lane >> 4, k0 = lh * 8;
  int gpix = blockIdx.x * 128 + wave * 16;
  int b = gpix >> 12;
  // GEMM1: [16pix x 256] @ [256 x 16] + fb1, relu
  float bb = fb1[m];
  v8f acc = {bb, bb, bb, bb, bb, bb, bb, bb};
  const _Float16* arow = med16 + ((size_t)gpix + m) * 256;
  for (int s = 0; s < 8; ++s) {
    v16h a = ld_frag2(arow + s * 32 + k0, arow + s * 32 + k0 + 16);
    v16h bm = ld_frag32(wpf1 + (size_t)s * 512 + lane * 16);
    acc = wmma32(a, bm, acc);
  }
  // zero K-pad region and write h (relu) into LDS in A-layout
#pragma unroll
  for (int j = 0; j < 8; ++j) hbuf[wave][m][16 + lh * 8 + j] = (_Float16)0.f;
#pragma unroll
  for (int v = 0; v < 8; ++v) hbuf[wave][v + 8 * lh][m] = (_Float16)fmaxf(acc[v], 0.f);
  __syncthreads();
  v16h a2 = ld_frag2(&hbuf[wave][m][k0], &hbuf[wave][m][k0 + 16]);
  int pixloc = (gpix & 4095) + lh * 8;
  for (int t = 0; t < 16; ++t) {
    int oc = t * 16 + m;
    float c0 = fb2[oc] + rv[b * 256 + oc];
    v8f acc2 = {c0, c0, c0, c0, c0, c0, c0, c0};
    v16h b2 = ld_frag32(wpf2 + (size_t)t * 512 + lane * 16);
    acc2 = wmma32(a2, b2, acc2);
    float* dst = out + ((size_t)b * 256 + oc) * 4096 + pixloc;
#pragma unroll
    for (int v = 0; v < 8; ++v) dst[v] = 1.f / (1.f + expf(-acc2[v]));
  }
}

// ---------- host launcher ----------
extern "C" void kernel_launch(void* const* d_in, const int* in_sizes, int n_in,
                              void* d_out, int out_size, void* d_ws, size_t ws_size,
                              hipStream_t stream) {
  const float* x   = (const float*)d_in[0];
  const float* w1  = (const float*)d_in[1];
  const float* b1  = (const float*)d_in[2];
  const float* w2  = (const float*)d_in[3];
  const float* b2  = (const float*)d_in[4];
  const float* g2  = (const float*)d_in[5];
  const float* bt2 = (const float*)d_in[6];
  const float* w3  = (const float*)d_in[7];
  const float* b3  = (const float*)d_in[8];
  const float* g3  = (const float*)d_in[9];
  const float* bt3 = (const float*)d_in[10];
  const float* w4  = (const float*)d_in[11];
  const float* b4  = (const float*)d_in[12];
  const float* g4  = (const float*)d_in[13];
  const float* bt4 = (const float*)d_in[14];
  const float* fw1 = (const float*)d_in[15];
  const float* fb1 = (const float*)d_in[16];
  const float* fw2 = (const float*)d_in[17];
  const float* fb2 = (const float*)d_in[18];
  float* out = (float*)d_out;

  char* p = (char*)d_ws;
  auto alloc = [&](size_t n) { void* r = (void*)p; p += (n + 255) & ~(size_t)255; return r; };
  _Float16* y16   = (_Float16*)alloc(8ull * 4096 * 64 * 2);   // conv1x1 out, NHWC f16
  float*    cat   = (float*)   alloc(8ull * 256 * 4096 * 4);  // concat buffer NCHW f32
  _Float16* med16 = (_Float16*)alloc(8ull * 4096 * 256 * 2);  // median out, NHWC f16
  _Float16* pool  = (_Float16*)alloc(8ull * 1024 * 64 * 2);   // maxpool out
  float*    stats = (float*)   alloc(384 * 4);                // [sum(192), sumsq(192)]
  float*    maxr  = (float*)   alloc(2048 * 4);
  float*    avgr  = (float*)   alloc(2048 * 4);
  float*    rvec  = (float*)   alloc(2048 * 4);
  _Float16* wp1   = (_Float16*)alloc(16384ull * 2);
  _Float16* wp3   = (_Float16*)alloc(9ull  * 4096 * 2);
  _Float16* wp5   = (_Float16*)alloc(25ull * 4096 * 2);
  _Float16* wp7   = (_Float16*)alloc(49ull * 4096 * 2);
  _Float16* wpf1  = (_Float16*)alloc(4096ull * 2);
  _Float16* wpf2  = (_Float16*)alloc(8192ull * 2);

  k_zero<<<2, 256, 0, stream>>>(stats, 384);
  k_pack_1x1 <<<64,  256, 0, stream>>>(w1, wp1);
  k_pack_conv<<<128, 256, 0, stream>>>(w2, wp3, 3);
  k_pack_conv<<<256, 256, 0, stream>>>(w3, wp5, 5);
  k_pack_conv<<<512, 256, 0, stream>>>(w4, wp7, 7);
  k_pack_fw1 <<<16,  256, 0, stream>>>(fw1, wpf1);
  k_pack_fw2 <<<32,  256, 0, stream>>>(fw2, wpf2);

  k_reduce  <<<2048, 256, 0, stream>>>(x, maxr, avgr);
  k_conv1x1 <<<512,  256, 0, stream>>>(x, wp1, b1, y16);
  k_conv<3> <<<512,  256, 0, stream>>>(y16, wp3, b2, cat, 0,   stats);
  k_conv<5> <<<512,  256, 0, stream>>>(y16, wp5, b3, cat, 64,  stats);
  k_conv<7> <<<512,  256, 0, stream>>>(y16, wp7, b4, cat, 128, stats);
  k_pool    <<<256,  256, 0, stream>>>(y16, pool);
  k_bilinear<<<1024, 256, 0, stream>>>(pool, cat);
  k_bnrelu  <<<24576, 256, 0, stream>>>(cat, stats, g2, bt2, g3, bt3, g4, bt4);
  k_median  <<<32768, 256, 0, stream>>>(cat, med16);
  k_rvec    <<<8,    256, 0, stream>>>(maxr, avgr, fw1, fb1, fw2, fb2, rvec);
  k_fc      <<<256,  256, 0, stream>>>(med16, wpf1, wpf2, fb1, fb2, rvec, out);

  (void)in_sizes; (void)n_in; (void)out_size; (void)ws_size;
}
